// MbHeadLayer_11458972745982
// MI455X (gfx1250) — matrix operand order repaired and checked by reading, not compiled, not run on hardware
//
#include <hip/hip_runtime.h>

typedef __attribute__((ext_vector_type(16))) __bf16 v16bf;
typedef __attribute__((ext_vector_type(8)))  __bf16 v8bf;
typedef __attribute__((ext_vector_type(8)))  float  v8f;

#define NMB    4
#define CCH    17
#define BATCH  8
#define HEADS  8
#define SEQ    512
#define DIM    64

// ---------------- workspace layout (offsets in bytes) ----------------
// [0,         2176)    : sm4  = float4[CCH][HEADS]   softmax(alpha1, axis=1)
// [4096,      +262144) : w1t  = bf16 [4][8][64][64]  (B,h,n,m)  == W1 transposed
// [266240,    +4194304): kbf  = bf16 (b,h,S,d)       key converted to bf16
#define WS_SM4 0
#define WS_W1T 4096
#define WS_KBF (4096 + 262144)

#define WMMA_BF16(A, B, C) \
  __builtin_amdgcn_wmma_f32_16x16x32_bf16(false, (A), false, (B), (short)0, (C), false, false)

// -------- tiny prep kernels --------

__global__ void softmax_kernel(const float* __restrict__ alpha1, float4* __restrict__ sm4) {
  int t = threadIdx.x;
  if (t >= CCH * HEADS) return;
  int c = t >> 3, h = t & 7;
  float v[NMB];
  float mx = -3.4e38f;
#pragma unroll
  for (int B = 0; B < NMB; ++B) {
    v[B] = alpha1[(c * NMB + B) * HEADS + h];
    mx = fmaxf(mx, v[B]);
  }
  float s = 0.f;
#pragma unroll
  for (int B = 0; B < NMB; ++B) { v[B] = __expf(v[B] - mx); s += v[B]; }
  float inv = 1.f / s;
  sm4[t] = make_float4(v[0] * inv, v[1] * inv, v[2] * inv, v[3] * inv);
}

// W1 (B,h,m,n) f32 -> w1t (B,h,n,m) bf16; output-coalesced
__global__ void w1t_kernel(const float* __restrict__ W1, __bf16* __restrict__ w1t) {
  int idx = blockIdx.x * 256 + threadIdx.x;        // 4*8*64*64 = 131072 total
  int m  = idx & 63;
  int n  = (idx >> 6) & 63;
  int bh = idx >> 12;                              // B*8 + h
  w1t[idx] = (__bf16)W1[(bh * 64 + m) * 64 + n];
}

__global__ void kconv_kernel(const float* __restrict__ k, __bf16* __restrict__ kbf) {
  int idx = blockIdx.x * 256 + threadIdx.x;        // 8*8*512*64 = 2097152 total
  kbf[idx] = (__bf16)k[idx];
}

// -------- main fused kernel --------
// grid = (b*h = 64, SEQ/64 = 8), block = 128 (4 waves). Each wave owns one
// 16-row i-tile: builds P_B = Q*W1[B] (WMMA, staged via LDS into A-operand
// layout), then sweeps all 32 j-tiles: G_B = P_B * K^T (WMMA) and combines
// with gathered softmax weights.
__global__ __launch_bounds__(128) void scores_kernel(
    const float*  __restrict__ q,     const int*   __restrict__ b_seq,
    const __bf16* __restrict__ w1t,   const __bf16* __restrict__ kbf,
    const float4* __restrict__ sm4,   float* __restrict__ out)
{
  __shared__ __align__(16) __bf16 p_lds[4][4][16][DIM];   // [wave][B][row][n] = 32 KB
  __shared__ float4 sm_lds[CCH * HEADS];                  // 2176 B

  const int tid  = threadIdx.x;
  const int wave = tid >> 5;
  const int lane = tid & 31;
  const int rl   = lane & 15;       // row (A/D) or column (B/D) within tile
  const int half = lane >> 4;       // which K-half this lane holds

  const int bh = blockIdx.x;        // b*8 + h
  const int bb = bh >> 3;
  const int hh = bh & 7;
  const int itile = blockIdx.y * 4 + wave;
  const int i0 = itile * 16;

  if (tid < CCH * HEADS) sm_lds[tid] = sm4[tid];

  // ---- load Q rows as bf16 A-operands (K = m dim, two 16x32 chunks) ----
  const float* qrow = q + ((size_t)bh * SEQ + i0 + rl) * DIM;
  v16bf aq[2];
#pragma unroll
  for (int kc = 0; kc < 2; ++kc) {
    const int base = kc * 32 + half * 16;
    float4 f0 = *(const float4*)(qrow + base);
    float4 f1 = *(const float4*)(qrow + base + 4);
    float4 f2 = *(const float4*)(qrow + base + 8);
    float4 f3 = *(const float4*)(qrow + base + 12);
    v16bf a;
    a[0]=(__bf16)f0.x;  a[1]=(__bf16)f0.y;  a[2]=(__bf16)f0.z;  a[3]=(__bf16)f0.w;
    a[4]=(__bf16)f1.x;  a[5]=(__bf16)f1.y;  a[6]=(__bf16)f1.z;  a[7]=(__bf16)f1.w;
    a[8]=(__bf16)f2.x;  a[9]=(__bf16)f2.y;  a[10]=(__bf16)f2.z; a[11]=(__bf16)f2.w;
    a[12]=(__bf16)f3.x; a[13]=(__bf16)f3.y; a[14]=(__bf16)f3.z; a[15]=(__bf16)f3.w;
    aq[kc] = a;
  }

  // ---- P stage: P_B = Q * W1[B,h]  (4 channels x 4 n-tiles x K=64) ----
#pragma unroll
  for (int Bc = 0; Bc < 4; ++Bc) {
#pragma unroll
    for (int nt = 0; nt < 4; ++nt) {
      const __bf16* wp = w1t + (((size_t)Bc * HEADS + hh) * DIM + nt * 16 + rl) * DIM
                             + half * 16;
      v16bf b0 = *(const v16bf*)(wp);        // m = 0..31  (this lane's half)
      v16bf b1 = *(const v16bf*)(wp + 32);   // m = 32..63
      v8f acc = {};
      acc = WMMA_BF16(aq[0], b0, acc);
      acc = WMMA_BF16(aq[1], b1, acc);
      // D layout: reg r -> row r + 8*half, col rl (within n-tile)
#pragma unroll
      for (int r = 0; r < 8; ++r)
        p_lds[wave][Bc][r + 8 * half][nt * 16 + rl] = (__bf16)acc[r];
    }
  }
  __syncthreads();

  // ---- re-read P as A-operands (K = n dim now) ----
  v16bf PA[4][2];
#pragma unroll
  for (int Bc = 0; Bc < 4; ++Bc) {
#pragma unroll
    for (int kc = 0; kc < 2; ++kc) {
      const __bf16* pp = &p_lds[wave][Bc][rl][kc * 32 + half * 16];
      v8bf lo = *(const v8bf*)(pp);
      v8bf hi = *(const v8bf*)(pp + 8);
      PA[Bc][kc] = __builtin_shufflevector(lo, hi,
          0,1,2,3,4,5,6,7,8,9,10,11,12,13,14,15);
    }
  }

  // per-lane row block ids (fixed for the whole j sweep)
  int bi_r[8];
#pragma unroll
  for (int r = 0; r < 8; ++r)
    bi_r[r] = b_seq[bb * SEQ + i0 + r + 8 * half];

  // ---- j sweep: G_B = P_B * K^T, fuse channel select + store ----
  const size_t outbase = (size_t)bh * SEQ * SEQ;
  const __bf16* kbase = kbf + (size_t)bh * SEQ * DIM;

  for (int jt = 0; jt < 32; ++jt) {
    const int j0 = jt * 16;
    const __bf16* kr = kbase + (size_t)(j0 + rl) * DIM + half * 16;
    v16bf kb0 = *(const v16bf*)(kr);        // n = 0..31 (this lane's half)
    v16bf kb1 = *(const v16bf*)(kr + 32);   // n = 32..63
    if (jt + 1 < 32) __builtin_prefetch(kr + 16 * DIM, 0, 0);  // next j-tile

    v8f a0 = {}, a1 = {}, a2 = {}, a3 = {};
    a0 = WMMA_BF16(PA[0][0], kb0, a0);  a0 = WMMA_BF16(PA[0][1], kb1, a0);
    a1 = WMMA_BF16(PA[1][0], kb0, a1);  a1 = WMMA_BF16(PA[1][1], kb1, a1);
    a2 = WMMA_BF16(PA[2][0], kb0, a2);  a2 = WMMA_BF16(PA[2][1], kb1, a2);
    a3 = WMMA_BF16(PA[3][0], kb0, a3);  a3 = WMMA_BF16(PA[3][1], kb1, a3);

    const int bj = b_seq[bb * SEQ + j0 + rl];   // this lane's column
#pragma unroll
    for (int r = 0; r < 8; ++r) {
      const int bi = bi_r[r];
      const int c  = (bi * bj != 0) ? ((bi - 1) * NMB + bj) : 0;
      const float4 w = sm_lds[c * HEADS + hh];
      const float sc = w.x * a0[r] + w.y * a1[r] + w.z * a2[r] + w.w * a3[r];
      out[outbase + (size_t)(i0 + r + 8 * half) * SEQ + (j0 + rl)] = sc;
    }
  }
}

extern "C" void kernel_launch(void* const* d_in, const int* in_sizes, int n_in,
                              void* d_out, int out_size, void* d_ws, size_t ws_size,
                              hipStream_t stream) {
  const float* q      = (const float*)d_in[0];
  const float* k      = (const float*)d_in[1];
  const int*   b_seq  = (const int*)  d_in[2];
  const float* W1     = (const float*)d_in[3];
  const float* alpha1 = (const float*)d_in[4];
  float* out = (float*)d_out;

  char* ws = (char*)d_ws;
  float4* sm4 = (float4*)(ws + WS_SM4);
  __bf16* w1t = (__bf16*)(ws + WS_W1T);
  __bf16* kbf = (__bf16*)(ws + WS_KBF);

  softmax_kernel<<<1, 160, 0, stream>>>(alpha1, sm4);
  w1t_kernel  <<<(4 * 8 * 64 * 64) / 256, 256, 0, stream>>>(W1, w1t);
  kconv_kernel<<<(BATCH * HEADS * SEQ * DIM) / 256, 256, 0, stream>>>(k, kbf);

  dim3 grid(BATCH * HEADS, SEQ / 64);
  scores_kernel<<<grid, 128, 0, stream>>>(q, b_seq, w1t, kbf, sm4, out);
}
